// DualSPRTLinear_15058155339869
// MI455X (gfx1250) — matrix operand order repaired and checked by reading, not compiled, not run on hardware
//
#include <hip/hip_runtime.h>
#include <hip/hip_bf16.h>

typedef __attribute__((ext_vector_type(16))) _Float16 v16h;
typedef __attribute__((ext_vector_type(8)))  _Float16 v8h;
typedef __attribute__((ext_vector_type(4)))  _Float16 v4h;
typedef __attribute__((ext_vector_type(8)))  float    v8f;

#define IN_F   4096
#define OUT_F  4096
#define GS     128
#define MTOT   8192

#define BM 128
#define BN 128
#define BK 128            // == GS, so one scale group per K tile
#define LDSP (BK + 8)     // padded LDS row stride in halves (16B pad)

__global__ __launch_bounds__(256)
void ternary_wmma_gemm(const float* __restrict__ X,        // [8192, 4096] f32
                       const signed char* __restrict__ W,  // [4096, 4096] int8 in {-1,0,1}
                       const float* __restrict__ S,        // [4096 * 32] group scales
                       float* __restrict__ O)              // [8192, 4096] f32
{
    __shared__ __align__(16) _Float16 As[BM * LDSP];  // x tile,   m-major
    __shared__ __align__(16) _Float16 Bs[BN * LDSP];  // ternary tile, n-major

    const int tid  = threadIdx.x;
    const int wave = tid >> 5;
    const int lane = tid & 31;
    const int lm   = lane & 15;   // row (A) / col (B,C) within 16
    const int lh   = lane >> 4;   // half-wave selector

    const int mBase = blockIdx.x * BM;
    const int nBase = blockIdx.y * BN;

    const int waveM = (wave >> 1) * 32;  // 0,32,64,96
    const int waveN = (wave & 1) * 64;   // 0,64

    v8f acc[2][4] = {};   // persistent f32 accumulators (scale applied)

    // staging assignments: 256 threads cover the tiles
    const int aRow = tid >> 1;          // 0..127
    const int aCol = (tid & 1) * 64;    // half-row of 64 floats
    const int bRow = tid >> 1;          // 0..127
    const int bCol = (tid & 1) * 64;    // half-row of 64 bytes

    const int nGroups = IN_F / BK;      // 32 K-groups
    for (int kt = 0; kt < nGroups; ++kt) {
        const int k0 = kt * BK;

        // ---- stage A: 128x128 f32 -> f16 into LDS (coalesced float4) ----
        {
            const float4* src = (const float4*)(X + (size_t)(mBase + aRow) * IN_F + k0 + aCol);
            _Float16* dst = &As[aRow * LDSP + aCol];
            #pragma unroll
            for (int i = 0; i < 16; ++i) {
                float4 v = src[i];
                v4h h;
                h.x = (_Float16)v.x; h.y = (_Float16)v.y;
                h.z = (_Float16)v.z; h.w = (_Float16)v.w;
                *(v4h*)(dst + i * 4) = h;
            }
            if (kt + 1 < nGroups)
                __builtin_prefetch((const void*)(src + 32), 0, 1);  // next K tile
        }

        // ---- stage B: 128x128 int8 ternary -> f16 into LDS ----
        {
            const int4* src = (const int4*)(W + (size_t)(nBase + bRow) * IN_F + k0 + bCol);
            _Float16* dst = &Bs[bRow * LDSP + bCol];
            #pragma unroll
            for (int i = 0; i < 4; ++i) {
                int4 p = src[i];
                int wd[4] = {p.x, p.y, p.z, p.w};
                #pragma unroll
                for (int j = 0; j < 4; ++j) {
                    int v = wd[j];
                    v4h h;
                    h.x = (_Float16)(float)(signed char)(v & 0xff);
                    h.y = (_Float16)(float)(signed char)((v >> 8) & 0xff);
                    h.z = (_Float16)(float)(signed char)((v >> 16) & 0xff);
                    h.w = (_Float16)(float)(signed char)(v >> 24);
                    *(v4h*)(dst + (i * 4 + j) * 4) = h;
                }
            }
            if (kt + 1 < nGroups)
                __builtin_prefetch((const void*)(src + 8), 0, 1);   // next K tile
        }
        __syncthreads();

        // ---- compute: 4 WMMA k-steps covering this 128-wide scale group ----
        v8f gacc[2][4] = {};   // unscaled group accumulators

        #pragma unroll
        for (int ks = 0; ks < BK; ks += 32) {
            // A fragments: 16x32 f16. lane holds row m=lm, K chunks
            // {lh*8 .. +7} and {lh*8+16 .. +23}  (ISA 16-bit A layout)
            v16h af[2];
            #pragma unroll
            for (int i = 0; i < 2; ++i) {
                const _Float16* ap = &As[(waveM + i * 16 + lm) * LDSP + ks + lh * 8];
                v8h lo = *(const v8h*)ap;          // K = ks+lh*8   .. +7
                v8h hi = *(const v8h*)(ap + 16);   // K = ks+lh*8+16.. +23
                #pragma unroll
                for (int e = 0; e < 8; ++e) { af[i][e] = lo[e]; af[i][e + 8] = hi[e]; }
            }
            // B fragments: 32x16 f16. lane holds col n=lm, K chunks
            // {lh*8 .. +7} and {lh*8+16 .. +23} (matches A layout)
            v16h bf[4];
            #pragma unroll
            for (int j = 0; j < 4; ++j) {
                const _Float16* bp = &Bs[(waveN + j * 16 + lm) * LDSP + ks + lh * 8];
                v8h lo = *(const v8h*)bp;
                v8h hi = *(const v8h*)(bp + 16);
                #pragma unroll
                for (int e = 0; e < 8; ++e) { bf[j][e] = lo[e]; bf[j][e + 8] = hi[e]; }
            }
            #pragma unroll
            for (int i = 0; i < 2; ++i)
                #pragma unroll
                for (int j = 0; j < 4; ++j)
                    gacc[i][j] = __builtin_amdgcn_wmma_f32_16x16x32_f16(
                        false, af[i], false, bf[j], (short)0, gacc[i][j], false, false);
        }

        // ---- fold group accumulators with per-(n, group) scale in f32 ----
        #pragma unroll
        for (int j = 0; j < 4; ++j) {
            const int n = nBase + waveN + j * 16 + lm;     // this lane's output column
            const float s = S[n * (IN_F / GS) + kt];
            #pragma unroll
            for (int i = 0; i < 2; ++i)
                #pragma unroll
                for (int e = 0; e < 8; ++e)
                    acc[i][j][e] += gacc[i][j][e] * s;
        }
        __syncthreads();   // protect LDS before next group's staging
    }

    // ---- store f32 output: D layout m = 8*lh + e, n = lm ----
    #pragma unroll
    for (int i = 0; i < 2; ++i)
        #pragma unroll
        for (int j = 0; j < 4; ++j) {
            const int n = nBase + waveN + j * 16 + lm;
            #pragma unroll
            for (int e = 0; e < 8; ++e) {
                const int m = mBase + waveM + i * 16 + lh * 8 + e;
                O[(size_t)m * OUT_F + n] = acc[i][j][e];
            }
        }
}

extern "C" void kernel_launch(void* const* d_in, const int* in_sizes, int n_in,
                              void* d_out, int out_size, void* d_ws, size_t ws_size,
                              hipStream_t stream) {
    (void)in_sizes; (void)n_in; (void)out_size; (void)d_ws; (void)ws_size;
    const float*       x = (const float*)d_in[0];
    const signed char* w = (const signed char*)d_in[1];   // int8 ternary codes
    const float*       s = (const float*)d_in[2];
    float*             o = (float*)d_out;

    dim3 grid(MTOT / BM, OUT_F / BN);   // (64, 32)
    ternary_wmma_gemm<<<grid, 256, 0, stream>>>(x, w, s, o);
}
